// QNetwork_63960652972282
// MI455X (gfx1250) — matrix operand order repaired and checked by reading, not compiled, not run on hardware
//
#include <hip/hip_runtime.h>

// ---------------------------------------------------------------------------
// GCN Q-network for MI455X (gfx1250, wave32, WMMA).
//  - GEMMs via v_wmma_f32_16x16x32_f16 with hi/lo f16 split (fp32-class accuracy)
//  - edge aggregation via global_atomic_add_f32 (L2-resident: h fits in 192MB L2)
//  - W pre-packed into the wave32 B-fragment layout for aligned 32B loads
// ---------------------------------------------------------------------------

#define NNODES 50000
#define DIM    256
#define ACTD   64

typedef _Float16 v16h __attribute__((ext_vector_type(16)));
typedef _Float16 v8h  __attribute__((ext_vector_type(8)));
typedef _Float16 v4h  __attribute__((ext_vector_type(4)));
typedef float    v8f  __attribute__((ext_vector_type(8)));

// ------------------------- degree / symmetric norm -------------------------
__global__ void k_deg_init(float* __restrict__ deg, int n) {
  int i = blockIdx.x * blockDim.x + threadIdx.x;
  if (i < n) deg[i] = 1.0f;                       // self-loop contributes 1
}

__global__ void k_deg_accum(const int* __restrict__ dst, float* __restrict__ deg, int nE) {
  int e = blockIdx.x * blockDim.x + threadIdx.x;
  if (e < nE) atomicAdd(&deg[dst[e]], 1.0f);
}

__global__ void k_dinv(float* __restrict__ deg, int n) {
  int i = blockIdx.x * blockDim.x + threadIdx.x;
  if (i < n) {
    float d = deg[i];
    deg[i] = (d > 0.0f) ? rsqrtf(d) : 0.0f;       // deg >= 1 always (self loops)
  }
}

// ---------------- pack W[k][n] (fp32) into B-fragment layout ---------------
// Layout per (col-tile ct, k-step ks, lane l, vgpr v):
//   n = ct*16 + (l&15);  k = ks*32 + (l>=16 ? 8 : 0) + (v>=4 ? 16 : 0) + 2(v&3) (+1 in hi half)
// hi plane at [0, 65536) halves, lo plane at [65536, 131072) halves.
__global__ void k_pack_w(const float* __restrict__ W, _Float16* __restrict__ out) {
  int t = blockIdx.x * blockDim.x + threadIdx.x;  // 0 .. 32767
  if (t >= 32768) return;
  int v  = t & 7;
  int l  = (t >> 3) & 31;
  int ks = (t >> 8) & 7;
  int ct = t >> 11;
  int n  = ct * 16 + (l & 15);
  int k0 = ks * 32 + ((l >> 4) << 3) + ((v >> 2) << 4) + 2 * (v & 3);
  float w0 = W[(size_t)k0 * DIM + n];
  float w1 = W[(size_t)(k0 + 1) * DIM + n];
  _Float16 h0 = (_Float16)w0, l0 = (_Float16)(w0 - (float)h0);
  _Float16 h1 = (_Float16)w1, l1 = (_Float16)(w1 - (float)h1);
  out[2 * t]             = h0;
  out[2 * t + 1]         = h1;
  out[65536 + 2 * t]     = l0;
  out[65536 + 2 * t + 1] = l1;
}

// --------------------- H = X @ W  via WMMA (hi/lo split) -------------------
__global__ void __launch_bounds__(256, 2)
k_gemm_wmma(const float* __restrict__ X, const _Float16* __restrict__ Wp,
            float* __restrict__ H, int nRows) {
  constexpr int LDA = DIM + 8;        // padded row stride (halves), 528B = 33*16B
  constexpr int ALO = 64 * LDA;       // lo-plane offset (halves)
  __shared__ _Float16 Alds[2 * 64 * LDA];

  const int tid     = threadIdx.x;
  const int rowBase = blockIdx.x * 64;

  // Stage 64x256 fp32 rows -> (hi,lo) f16 planes in LDS. Coalesced float4 reads.
  for (int i = 0; i < 16; ++i) {
    int fl  = tid + (i << 8);         // 0..4095 over 64 rows x 64 float4
    int r   = fl >> 6;
    int c4  = (fl & 63) << 2;
    int g   = rowBase + r;
    float4 xv = make_float4(0.f, 0.f, 0.f, 0.f);
    if (g < nRows) xv = *(const float4*)(X + (size_t)g * DIM + c4);
    v4h hi, lo;
    hi[0] = (_Float16)xv.x; lo[0] = (_Float16)(xv.x - (float)hi[0]);
    hi[1] = (_Float16)xv.y; lo[1] = (_Float16)(xv.y - (float)hi[1]);
    hi[2] = (_Float16)xv.z; lo[2] = (_Float16)(xv.z - (float)hi[2]);
    hi[3] = (_Float16)xv.w; lo[3] = (_Float16)(xv.w - (float)hi[3]);
    *(v4h*)&Alds[r * LDA + c4]       = hi;
    *(v4h*)&Alds[ALO + r * LDA + c4] = lo;
  }
  __syncthreads();

  const int lane  = tid & 31;
  const int w     = tid >> 5;         // wave 0..7
  const int khalf = (lane >> 4) << 3; // A layout: lanes 16-31 start at K+8
  const int mrow  = lane & 15;
  const int ct0   = w << 1;           // this wave owns col tiles ct0, ct0+1

  v8f acc[4][2] = {};

  for (int ks = 0; ks < 8; ++ks) {
    const int kb = ks << 5;
    const size_t b0off = (size_t)(((ct0) * 8 + ks) * 32 + lane) << 4;     // halves
    const size_t b1off = (size_t)(((ct0 + 1) * 8 + ks) * 32 + lane) << 4;
    v16h b0h = *(const v16h*)(Wp + b0off);
    v16h b1h = *(const v16h*)(Wp + b1off);
    v16h b0l = *(const v16h*)(Wp + 65536 + b0off);
    v16h b1l = *(const v16h*)(Wp + 65536 + b1off);

    for (int rt = 0; rt < 4; ++rt) {
      const _Float16* ap = &Alds[((rt << 4) + mrow) * LDA + kb + khalf];
      union { v16h v; v8h h[2]; } ah, al;
      ah.h[0] = *(const v8h*)(ap);            // K =  kb+khalf .. +7
      ah.h[1] = *(const v8h*)(ap + 16);       // K = +16 .. +23
      al.h[0] = *(const v8h*)(ap + ALO);
      al.h[1] = *(const v8h*)(ap + ALO + 16);

      // hi*hi + hi*lo + lo*hi  (lo*lo ~ 2^-22, dropped)
      acc[rt][0] = __builtin_amdgcn_wmma_f32_16x16x32_f16(false, ah.v, false, b0h,
                                                          (short)0, acc[rt][0], false, false);
      acc[rt][0] = __builtin_amdgcn_wmma_f32_16x16x32_f16(false, ah.v, false, b0l,
                                                          (short)0, acc[rt][0], false, false);
      acc[rt][0] = __builtin_amdgcn_wmma_f32_16x16x32_f16(false, al.v, false, b0h,
                                                          (short)0, acc[rt][0], false, false);
      acc[rt][1] = __builtin_amdgcn_wmma_f32_16x16x32_f16(false, ah.v, false, b1h,
                                                          (short)0, acc[rt][1], false, false);
      acc[rt][1] = __builtin_amdgcn_wmma_f32_16x16x32_f16(false, ah.v, false, b1l,
                                                          (short)0, acc[rt][1], false, false);
      acc[rt][1] = __builtin_amdgcn_wmma_f32_16x16x32_f16(false, al.v, false, b1h,
                                                          (short)0, acc[rt][1], false, false);
    }
  }

  // D layout: VGPR r -> M = r (lanes 0-15) / 8+r (lanes 16-31), N = lane&15
  const int mOff = (lane >> 4) << 3;
  const int col0 = (ct0 << 4) + (lane & 15);
  for (int rt = 0; rt < 4; ++rt)
    for (int r = 0; r < 8; ++r) {
      int row = rowBase + (rt << 4) + mOff + r;
      if (row < nRows) {
        H[(size_t)row * DIM + col0]      = acc[rt][0][r];
        H[(size_t)row * DIM + col0 + 16] = acc[rt][1][r];
      }
    }
}

// ------------------------- aggregation (self + edges) ----------------------
__global__ void k_self_init(const float* __restrict__ H, const float* __restrict__ dinv,
                            float* __restrict__ out) {
  int row = blockIdx.x;
  int c   = threadIdx.x;
  float di = dinv[row];
  size_t idx = (size_t)row * DIM + c;
  out[idx] = di * di * H[idx];
}

__global__ void k_edge_scatter(const float* __restrict__ H, const int* __restrict__ src,
                               const int* __restrict__ dst, const float* __restrict__ dinv,
                               float* __restrict__ out, int nE) {
  int t  = blockIdx.x * blockDim.x + threadIdx.x;
  int e  = t >> 6;                    // 64 groups of 4 channels per edge
  if (e >= nE) return;
  int cg = (t & 63) << 2;
  int s = src[e], d = dst[e];
  float w = dinv[s] * dinv[d];
  float4 h4 = *(const float4*)(H + (size_t)s * DIM + cg);
  float* o = out + (size_t)d * DIM + cg;
  atomicAdd(o + 0, w * h4.x);
  atomicAdd(o + 1, w * h4.y);
  atomicAdd(o + 2, w * h4.z);
  atomicAdd(o + 3, w * h4.w);
}

__global__ void k_bias_relu(float* __restrict__ h, const float* __restrict__ b) {
  size_t idx = (size_t)blockIdx.x * DIM + threadIdx.x;
  float v = h[idx] + b[threadIdx.x];
  h[idx] = v > 0.0f ? v : 0.0f;
}

// ------------------------------ head -------------------------------------
__global__ void k_zero256(float* __restrict__ g) { g[threadIdx.x] = 0.0f; }

__global__ void k_colsum(const float* __restrict__ h, float* __restrict__ gacc, int nRows) {
  int c  = threadIdx.x;
  int r0 = blockIdx.x << 8;
  int r1 = r0 + 256; if (r1 > nRows) r1 = nRows;
  float s = 0.0f;
  for (int r = r0; r < r1; ++r) s += h[(size_t)r * DIM + c];
  atomicAdd(&gacc[c], s);
}

__global__ void k_fc_head(const float* __restrict__ gacc,
                          const float* __restrict__ W1, const float* __restrict__ b1,
                          const float* __restrict__ W2, const float* __restrict__ b2,
                          const float* __restrict__ W3, const float* __restrict__ b3,
                          float* __restrict__ out, int nNodes) {
  __shared__ float g[DIM];
  int t = threadIdx.x;
  float m = gacc[t] * (1.0f / (float)nNodes);
  g[t] = m > 0.0f ? m : 0.0f;                      // relu(mean)
  __syncthreads();
  float a = b1[t];
  for (int k = 0; k < DIM; ++k) a += g[k] * W1[(size_t)k * DIM + t];
  a = a > 0.0f ? a : 0.0f;
  __syncthreads(); g[t] = a; __syncthreads();
  a = b2[t];
  for (int k = 0; k < DIM; ++k) a += g[k] * W2[(size_t)k * DIM + t];
  a = a > 0.0f ? a : 0.0f;
  __syncthreads(); g[t] = a; __syncthreads();
  if (t < ACTD) {
    float o = b3[t];
    for (int k = 0; k < DIM; ++k) o += g[k] * W3[(size_t)k * ACTD + t];
    out[t] = o;
  }
}

// ------------------------------ launch -------------------------------------
extern "C" void kernel_launch(void* const* d_in, const int* in_sizes, int n_in,
                              void* d_out, int out_size, void* d_ws, size_t ws_size,
                              hipStream_t stream) {
  const float* x    = (const float*)d_in[0];
  const int*   ei   = (const int*)d_in[1];
  const float* W1   = (const float*)d_in[2];
  const float* b1   = (const float*)d_in[3];
  const float* W2   = (const float*)d_in[4];
  const float* b2   = (const float*)d_in[5];
  const float* fcW1 = (const float*)d_in[6];
  const float* fcb1 = (const float*)d_in[7];
  const float* fcW2 = (const float*)d_in[8];
  const float* fcb2 = (const float*)d_in[9];
  const float* fcW3 = (const float*)d_in[10];
  const float* fcb3 = (const float*)d_in[11];
  float* out = (float*)d_out;

  const int nE = in_sizes[1] / 2;
  const int* src = ei;
  const int* dst = ei + nE;

  // Workspace layout (in floats):
  //   [0,        65536)  Wpack1 (hi+lo f16 planes, 256KB)
  //   [65536,   131072)  Wpack2
  //   [131072,  181120)  dinv (50000 + pad)
  //   [181120,  181376)  gacc (256)
  //   [181376, 12981376) B1  (50000 x 256)  gemm output
  //   [12981376, 25781376) B2 (50000 x 256) aggregation / layer output
  const size_t need = (size_t)25781376 * sizeof(float);
  if (ws_size < need) return;

  float*    wsf  = (float*)d_ws;
  _Float16* Wp1  = (_Float16*)(wsf);
  _Float16* Wp2  = (_Float16*)(wsf + 65536);
  float*    dinv = wsf + 131072;
  float*    gacc = wsf + 181120;
  float*    B1   = wsf + 181376;
  float*    B2   = wsf + 12981376;

  const int T = 256;
  const int gemmBlocks = (NNODES + 63) / 64;     // 782 (last block partially valid)
  const int edgeBlocks = (nE + 3) / 4;           // 64 threads/edge, 4 edges/block

  // norm
  k_deg_init <<<(NNODES + T - 1) / T, T, 0, stream>>>(dinv, NNODES);
  k_deg_accum<<<(nE + T - 1) / T,     T, 0, stream>>>(dst, dinv, nE);
  k_dinv     <<<(NNODES + T - 1) / T, T, 0, stream>>>(dinv, NNODES);

  // weight packing
  k_pack_w<<<128, T, 0, stream>>>(W1, Wp1);
  k_pack_w<<<128, T, 0, stream>>>(W2, Wp2);

  // layer 1
  k_gemm_wmma   <<<gemmBlocks, T, 0, stream>>>(x, Wp1, B1, NNODES);
  k_self_init   <<<NNODES, T, 0, stream>>>(B1, dinv, B2);
  k_edge_scatter<<<edgeBlocks, T, 0, stream>>>(B1, src, dst, dinv, B2, nE);
  k_bias_relu   <<<NNODES, T, 0, stream>>>(B2, b1);

  // layer 2 (B2 -> gemm -> B1 -> aggregate back into B2)
  k_gemm_wmma   <<<gemmBlocks, T, 0, stream>>>(B2, Wp2, B1, NNODES);
  k_self_init   <<<NNODES, T, 0, stream>>>(B1, dinv, B2);
  k_edge_scatter<<<edgeBlocks, T, 0, stream>>>(B1, src, dst, dinv, B2, nE);
  k_bias_relu   <<<NNODES, T, 0, stream>>>(B2, b2);

  // head
  k_zero256<<<1, T, 0, stream>>>(gacc);
  k_colsum <<<(NNODES + 255) / 256, T, 0, stream>>>(B2, gacc, NNODES);
  k_fc_head<<<1, T, 0, stream>>>(gacc, fcW1, fcb1, fcW2, fcb2, fcW3, fcb3, out, NNODES);
}
